// WindowAttention_33578054320257
// MI455X (gfx1250) — matrix operand order repaired and checked by reading, not compiled, not run on hardware
//
#include <hip/hip_runtime.h>
#include <hip/hip_bf16.h>

// ---- WMMA fragment types (wave32, CDNA5) -------------------------------
typedef __attribute__((ext_vector_type(16))) __bf16 bfrag16;  // A or B, 16x16x32 bf16
typedef __attribute__((ext_vector_type(8)))  __bf16 bvec8;
typedef __attribute__((ext_vector_type(8)))  float  ffrag8;   // C/D 16x16 f32

static __device__ __forceinline__ ffrag8 zero8() {
    ffrag8 z;
#pragma unroll
    for (int i = 0; i < 8; ++i) z[i] = 0.0f;
    return z;
}

static __device__ __forceinline__ bvec8 ld8(const __bf16* p) {
    return *(const bvec8*)p;   // 16B aligned ds_load_b128
}

static __device__ __forceinline__ bfrag16 pack16(bvec8 lo, bvec8 hi) {
    bfrag16 r;
#pragma unroll
    for (int i = 0; i < 8; ++i) { r[i] = lo[i]; r[i + 8] = hi[i]; }
    return r;
}

#define WMMA_BF16(A, B, C) \
    __builtin_amdgcn_wmma_f32_16x16x32_bf16(false, (A), false, (B), (short)0, (C), false, false)

// ---- CDNA5 async global->LDS copy (ASYNCcnt-tracked, no VGPR bounce) ----
static __device__ __forceinline__ void async_ld_b128(unsigned lds_byte_addr,
                                                     unsigned gl_byte_off,
                                                     const void* sbase) {
    // GVS mode: mem = SADDR + VGPR_off; dest LDS byte address in VGPR.
    asm volatile("global_load_async_to_lds_b128 %0, %1, %2"
                 :
                 : "v"(lds_byte_addr), "v"(gl_byte_off), "s"(sbase)
                 : "memory");
}
static __device__ __forceinline__ void wait_async0() {
    asm volatile("s_wait_asynccnt 0" ::: "memory");
}

// ---- problem constants --------------------------------------------------
// B_=2048, N=49, C=512, H=16, hd=32, nW=64, M = B_*N = 100352
#define MTOT   100352
#define CDIM   512
#define HEADS  16
#define HD     32
#define NTOK   49
#define NPAD   64
#define NWIN   64
#define SCALE  0.17677669529663687f   // 32^-0.5

// ========================================================================
// Kernel 1: qkv = x @ qkv_w^T + qkv_b  ->  scatter q*scale / k / v as bf16
// tile: 128 (M) x 64 (N), 8 waves, K-steps of 32.
// Grid: (n-blocks, m-blocks) so the 24 column tiles of one A-panel run
// back-to-back -> x streamed from HBM once, weights L2-resident.
// ========================================================================
__global__ __launch_bounds__(256) void qkv_gemm_kernel(
    const float* __restrict__ x,     // (100352, 512)
    const float* __restrict__ w,     // (1536, 512)
    const float* __restrict__ bias,  // (1536,)
    __bf16* __restrict__ qb, __bf16* __restrict__ kb, __bf16* __restrict__ vb)
{
    __shared__ __align__(16) __bf16 As[128][40];
    __shared__ __align__(16) __bf16 Bs[64][40];

    const int n0   = blockIdx.x * 64;
    const int m0   = blockIdx.y * 128;
    const int tid  = threadIdx.x;
    const int lane = tid & 31, wv = tid >> 5;
    const int l15  = lane & 15, hh = lane >> 4;

    ffrag8 acc[4];
#pragma unroll
    for (int nt = 0; nt < 4; ++nt) acc[nt] = zero8();

    for (int k0 = 0; k0 < CDIM; k0 += 32) {
        // ---- issue ALL panel loads first (distinct regs -> full MLP),
        //      then convert + store to LDS ----
        float4 fa[4], fb[2];
#pragma unroll
        for (int i = 0; i < 4; ++i) {
            const int idx = tid + i * 256;
            const int r = idx >> 3, kc = (idx & 7) * 4;
            fa[i] = *(const float4*)(x + (size_t)(m0 + r) * CDIM + k0 + kc);
        }
#pragma unroll
        for (int i = 0; i < 2; ++i) {
            const int idx = tid + i * 256;
            const int r = idx >> 3, kc = (idx & 7) * 4;
            fb[i] = *(const float4*)(w + (size_t)(n0 + r) * CDIM + k0 + kc);
        }
        // prefetch next K panel (global_prefetch_b8)
        if (k0 + 32 < CDIM) {
            __builtin_prefetch(x + (size_t)(m0 + (tid >> 1)) * CDIM + k0 + 32 + (tid & 1) * 16, 0, 0);
            if (tid < 128)
                __builtin_prefetch(w + (size_t)(n0 + (tid >> 1)) * CDIM + k0 + 32 + (tid & 1) * 16, 0, 0);
        }
#pragma unroll
        for (int i = 0; i < 4; ++i) {
            const int idx = tid + i * 256;
            const int r = idx >> 3, kc = (idx & 7) * 4;
            As[r][kc + 0] = (__bf16)fa[i].x; As[r][kc + 1] = (__bf16)fa[i].y;
            As[r][kc + 2] = (__bf16)fa[i].z; As[r][kc + 3] = (__bf16)fa[i].w;
        }
#pragma unroll
        for (int i = 0; i < 2; ++i) {
            const int idx = tid + i * 256;
            const int r = idx >> 3, kc = (idx & 7) * 4;
            Bs[r][kc + 0] = (__bf16)fb[i].x; Bs[r][kc + 1] = (__bf16)fb[i].y;
            Bs[r][kc + 2] = (__bf16)fb[i].z; Bs[r][kc + 3] = (__bf16)fb[i].w;
        }
        __syncthreads();

        const bfrag16 af = pack16(ld8(&As[wv * 16 + l15][hh * 8]),
                                  ld8(&As[wv * 16 + l15][16 + hh * 8]));
#pragma unroll
        for (int nt = 0; nt < 4; ++nt) {
            const bfrag16 bf = pack16(ld8(&Bs[nt * 16 + l15][hh * 8]),
                                      ld8(&Bs[nt * 16 + l15][16 + hh * 8]));
            acc[nt] = WMMA_BF16(af, bf, acc[nt]);
        }
        __syncthreads();
    }

    // epilogue: +bias, scale q, scatter to (B, H, N, hd) bf16
    const int which = n0 >> 9;          // 0=q 1=k 2=v (whole 64-col tile in one)
    const int cbase = n0 & 511;
    __bf16* dst = (which == 0) ? qb : ((which == 1) ? kb : vb);
    const float mul = (which == 0) ? SCALE : 1.0f;
#pragma unroll
    for (int nt = 0; nt < 4; ++nt) {
        const int c = cbase + nt * 16 + l15;
        const int head = c >> 5, d = c & 31;
        const float bv = bias[n0 + nt * 16 + l15];
#pragma unroll
        for (int r = 0; r < 8; ++r) {
            const int m = m0 + wv * 16 + r + hh * 8;  // < 100352 always (784*128)
            const int b = m / NTOK, t = m - b * NTOK;
            const float v = (acc[nt][r] + bv) * mul;
            dst[(((size_t)b * HEADS + head) * NTOK + t) * HD + d] = (__bf16)v;
        }
    }
}

// ========================================================================
// Kernel 2: per (b, h): S = q k^T (+relpos bias +mask), softmax, O = P v
// 4 waves, each owns a 16-row slab of the 64 (padded) tokens.
// Q/K staged with global_load_async_to_lds_b128 (pitch 56 = 112B: 16B
// aligned rows + 16 distinct bank groups for fragment reads).
// ========================================================================
__global__ __launch_bounds__(128) void window_attn_kernel(
    const __bf16* __restrict__ qb, const __bf16* __restrict__ kb,
    const __bf16* __restrict__ vb,
    const float* __restrict__ bias_table,  // (169, 16)
    const float* __restrict__ mask,        // (64, 49, 49)
    __bf16* __restrict__ ob)               // (B, N, C) bf16
{
    __shared__ __align__(16) __bf16 Ql[64][56];
    __shared__ __align__(16) __bf16 Kl[64][56];
    __shared__ __align__(16) __bf16 Vl[32][72];   // transposed: [d][token]
    __shared__ __align__(16) __bf16 Pl[64][72];

    const int b = blockIdx.x, h = blockIdx.y;
    const int tid = threadIdx.x, lane = tid & 31, wv = tid >> 5;
    const int l15 = lane & 15, hh = lane >> 4;
    const size_t base = ((size_t)b * HEADS + h) * NTOK * HD;

    // ---- async stage Q, K: 49 rows x 4 x 16B chunks each ----
    {
        const unsigned qlds = (unsigned)(size_t)&Ql[0][0];
        const unsigned klds = (unsigned)(size_t)&Kl[0][0];
        const __bf16* qg = qb + base;
        const __bf16* kg = kb + base;
        for (int i = tid; i < NTOK * 4; i += 128) {
            const int t = i >> 2, c = (i & 3) * 8;            // 8 bf16 = 16B
            const unsigned goff = (unsigned)((t * HD + c) * 2);
            const unsigned loff = (unsigned)((t * 56 + c) * 2);
            async_ld_b128(qlds + loff, goff, qg);
            async_ld_b128(klds + loff, goff, kg);
        }
    }
    // ---- V: vectorized loads, transpose-scatter into LDS ----
    {
        bvec8 vr[2];
#pragma unroll
        for (int i = 0; i < 2; ++i) {
            const int idx = tid + i * 128;                     // 0..255
            const int t = idx >> 2, c = (idx & 3) * 8;
            if (t < NTOK) {
                vr[i] = ld8(vb + base + t * HD + c);
            } else {
#pragma unroll
                for (int j = 0; j < 8; ++j) vr[i][j] = (__bf16)0.f;
            }
        }
#pragma unroll
        for (int i = 0; i < 2; ++i) {
            const int idx = tid + i * 128;
            const int t = idx >> 2, c = (idx & 3) * 8;
#pragma unroll
            for (int j = 0; j < 8; ++j) Vl[c + j][t] = vr[i][j];
        }
    }
    // ---- zero-pad Q/K rows 49..63 ----
    for (int e = tid; e < (NPAD - NTOK) * HD; e += 128) {
        const int t = NTOK + (e >> 5), d = e & 31;
        Ql[t][d] = (__bf16)0.f;
        Kl[t][d] = (__bf16)0.f;
    }
    wait_async0();
    __syncthreads();

    // ---- S = q k^T : K-dim = hd = 32 -> single WMMA per 16x16 tile ----
    ffrag8 s[4];
#pragma unroll
    for (int nt = 0; nt < 4; ++nt) s[nt] = zero8();
    {
        const bfrag16 af = pack16(ld8(&Ql[wv * 16 + l15][hh * 8]),
                                  ld8(&Ql[wv * 16 + l15][16 + hh * 8]));
#pragma unroll
        for (int nt = 0; nt < 4; ++nt) {
            const bfrag16 bf = pack16(ld8(&Kl[nt * 16 + l15][hh * 8]),
                                      ld8(&Kl[nt * 16 + l15][16 + hh * 8]));
            s[nt] = WMMA_BF16(af, bf, s[nt]);
        }
    }

    // ---- relative position bias + window mask + column padding ----
    const int widx = b & (NWIN - 1);
#pragma unroll
    for (int nt = 0; nt < 4; ++nt) {
        const int n = nt * 16 + l15;
#pragma unroll
        for (int r = 0; r < 8; ++r) {
            const int m = wv * 16 + r + hh * 8;
            if (n >= NTOK) {
                s[nt][r] = -1e30f;
            } else if (m < NTOK) {
                const int mi = m / 7, mj = m - mi * 7;
                const int ni = n / 7, nj = n - ni * 7;
                const int ridx = (mi - ni + 6) * 13 + (mj - nj + 6);
                s[nt][r] += bias_table[ridx * HEADS + h] +
                            mask[((size_t)widx * NTOK + m) * NTOK + n];
            }
        }
    }

    // ---- row softmax: row m lives in 16 lanes of one half-wave ----
#pragma unroll
    for (int r = 0; r < 8; ++r) {
        const float a0 = s[0][r], a1 = s[1][r], a2 = s[2][r], a3 = s[3][r];
        float mx = fmaxf(fmaxf(a0, a1), fmaxf(a2, a3));
#pragma unroll
        for (int off = 1; off < 16; off <<= 1)
            mx = fmaxf(mx, __shfl_xor(mx, off, 32));
        const float e0 = __expf(a0 - mx), e1 = __expf(a1 - mx);
        const float e2 = __expf(a2 - mx), e3 = __expf(a3 - mx);
        float sum = e0 + e1 + e2 + e3;
#pragma unroll
        for (int off = 1; off < 16; off <<= 1)
            sum += __shfl_xor(sum, off, 32);
        const float inv = 1.0f / sum;
        const int m = wv * 16 + r + hh * 8;
        Pl[m][ 0 + l15] = (__bf16)(e0 * inv);
        Pl[m][16 + l15] = (__bf16)(e1 * inv);
        Pl[m][32 + l15] = (__bf16)(e2 * inv);
        Pl[m][48 + l15] = (__bf16)(e3 * inv);
    }
    __syncthreads();

    // ---- O = P @ V : K = 64 (2 k-steps), N = hd = 32 (2 tiles) ----
    ffrag8 o[2];
#pragma unroll
    for (int nt = 0; nt < 2; ++nt) o[nt] = zero8();
#pragma unroll
    for (int ks = 0; ks < 2; ++ks) {
        const int k0 = ks * 32;
        const bfrag16 a2 = pack16(ld8(&Pl[wv * 16 + l15][k0 + hh * 8]),
                                  ld8(&Pl[wv * 16 + l15][k0 + 16 + hh * 8]));
#pragma unroll
        for (int nt = 0; nt < 2; ++nt) {
            const int d = nt * 16 + l15;
            const bfrag16 b2 = pack16(ld8(&Vl[d][k0 + hh * 8]),
                                      ld8(&Vl[d][k0 + 16 + hh * 8]));
            o[nt] = WMMA_BF16(a2, b2, o[nt]);
        }
    }

    // scatter to (B, N, C) bf16
#pragma unroll
    for (int nt = 0; nt < 2; ++nt) {
#pragma unroll
        for (int r = 0; r < 8; ++r) {
            const int t = wv * 16 + r + hh * 8;
            if (t < NTOK) {
                const int c = h * HD + nt * 16 + l15;
                ob[((size_t)b * NTOK + t) * CDIM + c] = (__bf16)o[nt][r];
            }
        }
    }
}

// ========================================================================
// Kernel 3: out = y @ proj_w^T + proj_b   (bf16 A async->LDS, f32 B
// converted through VGPRs, f32 out).  Grid (n-blocks, m-blocks) for L2 reuse.
// ========================================================================
__global__ __launch_bounds__(256) void proj_gemm_kernel(
    const __bf16* __restrict__ a,    // (100352, 512) bf16
    const float* __restrict__ w,     // (512, 512)
    const float* __restrict__ bias,  // (512,)
    float* __restrict__ out)         // (100352, 512)
{
    __shared__ __align__(16) __bf16 As[128][56];  // async-staged: 16B rows
    __shared__ __align__(16) __bf16 Bs[64][40];

    const int n0   = blockIdx.x * 64;
    const int m0   = blockIdx.y * 128;
    const int tid  = threadIdx.x;
    const int lane = tid & 31, wv = tid >> 5;
    const int l15  = lane & 15, hh = lane >> 4;
    const unsigned alds = (unsigned)(size_t)&As[0][0];

    ffrag8 acc[4];
#pragma unroll
    for (int nt = 0; nt < 4; ++nt) acc[nt] = zero8();

    for (int k0 = 0; k0 < CDIM; k0 += 32) {
        // stage A: 128 rows x 4 x 16B chunks, async global->LDS (bf16, no cvt)
#pragma unroll
        for (int i = 0; i < 2; ++i) {
            const int idx = tid + i * 256;           // 0..511
            const int r = idx >> 2, c = (idx & 3) * 8;
            const unsigned goff = (unsigned)(((size_t)(m0 + r) * CDIM + k0 + c) * 2);
            const unsigned loff = (unsigned)((r * 56 + c) * 2);
            async_ld_b128(alds + loff, goff, a);
        }
        // stage B: 64x32 f32 -> bf16 (loads first, then convert)
        float4 fb[2];
#pragma unroll
        for (int i = 0; i < 2; ++i) {
            const int idx = tid + i * 256;
            const int r = idx >> 3, kc = (idx & 7) * 4;
            fb[i] = *(const float4*)(w + (size_t)(n0 + r) * CDIM + k0 + kc);
        }
        if (k0 + 32 < CDIM)
            __builtin_prefetch(a + (size_t)(m0 + (tid >> 1)) * CDIM + k0 + 32 + (tid & 1) * 16, 0, 0);
#pragma unroll
        for (int i = 0; i < 2; ++i) {
            const int idx = tid + i * 256;
            const int r = idx >> 3, kc = (idx & 7) * 4;
            Bs[r][kc + 0] = (__bf16)fb[i].x; Bs[r][kc + 1] = (__bf16)fb[i].y;
            Bs[r][kc + 2] = (__bf16)fb[i].z; Bs[r][kc + 3] = (__bf16)fb[i].w;
        }
        wait_async0();
        __syncthreads();

        const bfrag16 af = pack16(ld8(&As[wv * 16 + l15][hh * 8]),
                                  ld8(&As[wv * 16 + l15][16 + hh * 8]));
#pragma unroll
        for (int nt = 0; nt < 4; ++nt) {
            const bfrag16 bf = pack16(ld8(&Bs[nt * 16 + l15][hh * 8]),
                                      ld8(&Bs[nt * 16 + l15][16 + hh * 8]));
            acc[nt] = WMMA_BF16(af, bf, acc[nt]);
        }
        __syncthreads();
    }

#pragma unroll
    for (int nt = 0; nt < 4; ++nt) {
        const int n = n0 + nt * 16 + l15;
        const float bv = bias[n];
#pragma unroll
        for (int r = 0; r < 8; ++r) {
            const int m = m0 + wv * 16 + r + hh * 8;
            out[(size_t)m * CDIM + n] = acc[nt][r] + bv;
        }
    }
}

// ========================================================================
extern "C" void kernel_launch(void* const* d_in, const int* in_sizes, int n_in,
                              void* d_out, int out_size, void* d_ws, size_t ws_size,
                              hipStream_t stream) {
    const float* x          = (const float*)d_in[0];
    const float* mask       = (const float*)d_in[1];
    const float* qkv_w      = (const float*)d_in[2];
    const float* qkv_b      = (const float*)d_in[3];
    const float* proj_w     = (const float*)d_in[4];
    const float* proj_b     = (const float*)d_in[5];
    const float* bias_table = (const float*)d_in[6];
    float* out = (float*)d_out;

    const size_t QKV_ELEMS = (size_t)2048 * HEADS * NTOK * HD;  // 51,380,224
    __bf16* qb = (__bf16*)d_ws;
    __bf16* kb = qb + QKV_ELEMS;
    __bf16* vb = kb + QKV_ELEMS;
    __bf16* ob = vb + QKV_ELEMS;   // (B, N, C) bf16

    // grid = (n-blocks, m-blocks): consecutive WGs share one A-panel -> x/ob
    // read from HBM once; weight panels stay L2-resident.
    qkv_gemm_kernel<<<dim3((3 * CDIM) / 64, MTOT / 128), 256, 0, stream>>>(
        x, qkv_w, qkv_b, qb, kb, vb);

    window_attn_kernel<<<dim3(2048, HEADS), 128, 0, stream>>>(
        qb, kb, vb, bias_table, mask, ob);

    proj_gemm_kernel<<<dim3(CDIM / 64, MTOT / 128), 256, 0, stream>>>(
        ob, proj_w, proj_b, out);
}
